// HMModel_GAttN_MEAN_GRU_Att_DOT_88828513616043
// MI455X (gfx1250) — matrix operand order repaired and checked by reading, not compiled, not run on hardware
//
#include <hip/hip_runtime.h>
#include <cmath>

// ---------------------------------------------------------------------------
// GAT-style edge attention + aggregation for MI455X (gfx1250, wave32, WMMA).
//   D = 128 features, K = 2D = 256 concat dim.
//   Edge MLP + score + segment-max fused into one edge pass (WMMA bf16).
//   Softmax normalization deferred into the final GEMM (divide-by-denom on
//   the A-operand load) so edges are only touched twice total.
//   Packed bf16 weight fragments staged in LDS (320KB/WGP on CDNA5) so the
//   per-tile B-operand traffic is WGP-local instead of hammering L2.
//   A compiler memory barrier inside the persistent tile loop keeps the LDS
//   B-fragment loads in-loop (prevents hoist -> VGPR blowup -> scratch spill).
// ---------------------------------------------------------------------------

typedef __attribute__((ext_vector_type(16))) __bf16 v16bf;
typedef __attribute__((ext_vector_type(8)))  float  v8f;

#define DIM   128
#define TWO_D 256
#define WFRAG_BYTES (64 * 1024)          // 64 fragments x 32 lanes x 16 bf16
#define WFRAG_BF16  (WFRAG_BYTES / 2)
#define WFRAG_INT4  (WFRAG_BYTES / 16)

__device__ __forceinline__ void atomic_max_f32(float* addr, float val) {
  // Classic signed/unsigned split trick; correct for mixed signs with -inf init.
  if (val >= 0.0f)
    atomicMax((int*)addr, __float_as_int(val));
  else
    atomicMin((unsigned int*)addr, (unsigned int)__float_as_int(val));
}

__device__ __forceinline__ void atomic_add_f32(float* addr, float v) {
  __hip_atomic_fetch_add(addr, v, __ATOMIC_RELAXED, __HIP_MEMORY_SCOPE_AGENT);
}

// ---------------------------------------------------------------------------
// Kernel 0: pack W (row-major [128 out][256 in], f32) into per-lane bf16 WMMA
// B-fragments. Fragment (kk, nt) covers K = kk*32..+31, N = nt*16..+15.
// B layout (16-bit, 32x16): lane n<16 -> N=n, elems = K {kb..kb+7, kb+16..kb+23};
//                           lane n>=16 -> N=n-16, elems = K {kb+8..kb+15, kb+24..kb+31}.
// Stored so a lane's 16 bf16 values are contiguous (one 32B load at runtime).
// ---------------------------------------------------------------------------
__global__ void k_pack_weights(const float* __restrict__ W1,
                               const float* __restrict__ Wfc,
                               __bf16* __restrict__ W1B,
                               __bf16* __restrict__ WfcB) {
  int tid = blockIdx.x * blockDim.x + threadIdx.x;   // 4096 threads total
  if (tid >= 4096) return;
  const float* W  = (tid < 2048) ? W1  : Wfc;
  __bf16*      Wb = (tid < 2048) ? W1B : WfcB;
  int t    = tid & 2047;
  int lane = t & 31;
  int f    = t >> 5;            // fragment index 0..63
  int kk   = f >> 3;            // K step
  int nt   = f & 7;             // N tile
  int N    = nt * 16 + (lane & 15);
  int kb   = kk * 32 + ((lane >> 4) ? 8 : 0);
  __bf16* dst = Wb + (size_t)t * 16;
#pragma unroll
  for (int i = 0; i < 16; ++i)
    dst[i] = (__bf16)W[(size_t)N * TWO_D + kb + i + (i >= 8 ? 8 : 0)];
}

// ---------------------------------------------------------------------------
// Kernel 1: init scratch: smax = -inf, denom = 0, neigh = 0.
// ---------------------------------------------------------------------------
__global__ void k_init(float* __restrict__ smax, float* __restrict__ denom,
                       float* __restrict__ neigh, int ndst) {
  long i = (long)blockIdx.x * blockDim.x + threadIdx.x;
  long total = (long)ndst * (DIM + 2);
  if (i >= total) return;
  if (i < ndst)               smax[i] = -INFINITY;
  else if (i < 2L * ndst)     denom[i - ndst] = 0.0f;
  else                        neigh[i - 2L * ndst] = 0.0f;
}

// ---------------------------------------------------------------------------
// Helper: cooperative 64KB global->LDS copy of the packed weight table
// (b128 granularity; one barrier; LDS is read-only afterwards).
// ---------------------------------------------------------------------------
__device__ __forceinline__ void stage_weights(__bf16* sW, const __bf16* gW) {
  const int4* s = (const int4*)gW;
  int4*       d = (int4*)sW;
  for (int i = threadIdx.x; i < WFRAG_INT4; i += 256)
    d[i] = s[i];
  __syncthreads();
}

// ---------------------------------------------------------------------------
// Helper: load this lane's A-fragment (16 bf16) for K-chunk kb of a 128-f32
// row, with optional scale. A layout (16-bit, 16x32):
//   lane<16: M=lane,    K = {kb..kb+7, kb+16..kb+23}
//   lane>=16: M=lane-16, K = {kb+8..kb+15, kb+24..kb+31}  (kb pre-offset by 8)
// ---------------------------------------------------------------------------
__device__ __forceinline__ v16bf load_a_frag(const float* src, int kb,
                                             float scale, bool valid) {
  float q[16];
  if (valid) {
    const float4* s4 = (const float4*)(src + kb);   // 32B aligned (kb % 8 == 0)
    ((float4*)q)[0] = s4[0];
    ((float4*)q)[1] = s4[1];
    ((float4*)q)[2] = s4[4];                        // src + kb + 16
    ((float4*)q)[3] = s4[5];
  } else {
#pragma unroll
    for (int i = 0; i < 16; ++i) q[i] = 0.0f;
  }
  v16bf a;
#pragma unroll
  for (int i = 0; i < 16; ++i) a[i] = (__bf16)(q[i] * scale);
  return a;
}

// ---------------------------------------------------------------------------
// Kernel 2: edge attention scores. One wave per 16-edge tile, persistent
// grid-stride over tiles so the LDS weight staging is amortized.
//   z = tanh([hd; m] @ W1^T + b1)   via 8x8 wmma_f32_16x16x32_bf16
//   s = z . W2 + b2                 via in-register dot + xor-shuffle reduce
//   score[e] = s;  atomicMax(smax[dst], s)
// ---------------------------------------------------------------------------
__global__ void __launch_bounds__(256)
k_edge_scores(const float* __restrict__ feat_src,
              const float* __restrict__ feat_dst,
              const int*   __restrict__ edge_src,
              const int*   __restrict__ edge_dst,
              const __bf16* __restrict__ W1B,
              const float* __restrict__ b1,
              const float* __restrict__ W2,
              const float* __restrict__ b2,
              float* __restrict__ score,
              float* __restrict__ smax, int E) {
  __shared__ __bf16 sW[WFRAG_BF16];
  stage_weights(sW, W1B);

  int lane   = threadIdx.x & 31;
  int wave0  = blockIdx.x * (blockDim.x >> 5) + (threadIdx.x >> 5);
  int nWaves = gridDim.x * (blockDim.x >> 5);
  int nTiles = (E + 15) / 16;
  int khalf  = (lane >> 4) ? 8 : 0;
  int ncol   = lane & 15;
  int mbase  = (lane >> 4) ? 8 : 0;

  // Preload per-lane epilogue constants into registers (the in-loop memory
  // clobber below would otherwise force them to reload every tile).
  float bj[8], wj[8];
#pragma unroll
  for (int nt = 0; nt < 8; ++nt) {
    bj[nt] = b1[nt * 16 + ncol];
    wj[nt] = W2[nt * 16 + ncol];
  }
  float b2v = b2[0];

  for (int tile = wave0; tile < nTiles; tile += nWaves) {
    // Compiler-only barrier: keep the LDS B-fragment loads inside the loop
    // (prevents invariant-hoisting of 2KB/lane of fragments -> scratch spill).
    asm volatile("" ::: "memory");

    long e0    = (long)tile * 16;
    int  eL    = (int)e0 + ncol;
    bool valid = eL < E;
    const float* hd = nullptr;
    const float* m  = nullptr;
    if (valid) {
      hd = feat_dst + (long)edge_dst[eL] * DIM;
      m  = feat_src + (long)edge_src[eL] * DIM;
    }

    v8f acc[8] = {};
    for (int kk = 0; kk < 8; ++kk) {
      const float* src = (kk < 4) ? hd : m;        // K<128: hd, K>=128: m
      int kb = (kk & 3) * 32 + khalf;
      v16bf a = load_a_frag(src, kb, 1.0f, valid);
      const __bf16* bbase = sW + (size_t)(kk * 256 + lane) * 16;
#pragma unroll
      for (int nt = 0; nt < 8; ++nt) {
        v16bf b = *(const v16bf*)(bbase + (size_t)nt * 512);
        acc[nt] = __builtin_amdgcn_wmma_f32_16x16x32_bf16(
            false, a, false, b, (short)0, acc[nt], false, false);
      }
    }

    // bias + tanh + dot with W2 (per-lane partial over its column j)
    float p[8];
#pragma unroll
    for (int r = 0; r < 8; ++r) p[r] = 0.0f;
#pragma unroll
    for (int nt = 0; nt < 8; ++nt) {
#pragma unroll
      for (int r = 0; r < 8; ++r)
        p[r] += tanhf(acc[nt][r] + bj[nt]) * wj[nt];
    }
    // reduce across the 16 lanes of each half (xor 1,2,4,8 stays in-half)
#pragma unroll
    for (int mask = 1; mask < 16; mask <<= 1)
#pragma unroll
      for (int r = 0; r < 8; ++r)
        p[r] += __shfl_xor(p[r], mask, 32);

    if (ncol == 0) {
#pragma unroll
      for (int r = 0; r < 8; ++r) {
        int e = (int)e0 + mbase + r;
        if (e < E) {
          float s = p[r] + b2v;
          score[e] = s;
          atomic_max_f32(&smax[edge_dst[e]], s);
        }
      }
    }
  }
}

// ---------------------------------------------------------------------------
// Kernel 3: per-edge scatter. One wave per edge.
//   ex = exp(score - smax[dst]);  denom[dst] += ex;  neigh[dst] += ex * m
// (normalization by denom happens in the final GEMM's A-load)
// ---------------------------------------------------------------------------
__global__ void k_edge_scatter(const float* __restrict__ feat_src,
                               const int*   __restrict__ edge_src,
                               const int*   __restrict__ edge_dst,
                               const float* __restrict__ score,
                               const float* __restrict__ smax,
                               float* __restrict__ denom,
                               float* __restrict__ neigh, int E) {
  int e    = blockIdx.x * (blockDim.x >> 5) + (threadIdx.x >> 5);
  int lane = threadIdx.x & 31;
  if (e >= E) return;
  int dst  = edge_dst[e];
  float ex = __expf(score[e] - smax[dst]);
  if (lane == 0) atomic_add_f32(&denom[dst], ex);
  const float4* m4 = (const float4*)(feat_src + (long)edge_src[e] * DIM);
  float4 mv = m4[lane];                              // 128 floats / 32 lanes
  float* nd = neigh + (long)dst * DIM + lane * 4;
  atomic_add_f32(nd + 0, ex * mv.x);
  atomic_add_f32(nd + 1, ex * mv.y);
  atomic_add_f32(nd + 2, ex * mv.z);
  atomic_add_f32(nd + 3, ex * mv.w);
}

// ---------------------------------------------------------------------------
// Kernel 4: out = relu([feat_dst | neigh/denom] @ Wfc^T + bfc).
// One wave per 16-row dst tile; LDS-staged weights shared by the block's
// 8 waves.
// ---------------------------------------------------------------------------
__global__ void __launch_bounds__(256)
k_final(const float* __restrict__ feat_dst,
        const float* __restrict__ neigh,
        const float* __restrict__ denom,
        const __bf16* __restrict__ WfcB,
        const float* __restrict__ bfc,
        float* __restrict__ out, int ndst) {
  __shared__ __bf16 sW[WFRAG_BF16];
  stage_weights(sW, WfcB);

  int wave = blockIdx.x * (blockDim.x >> 5) + (threadIdx.x >> 5);
  int lane = threadIdx.x & 31;
  long r0 = (long)wave * 16;
  if (r0 >= ndst) return;

  int  row   = (int)r0 + (lane & 15);
  bool valid = row < ndst;
  float inv = 0.0f;
  const float* hs = nullptr;
  const float* na = nullptr;
  if (valid) {
    float dn = denom[row];
    inv = (dn > 0.0f) ? (1.0f / dn) : 0.0f;          // 0-degree nodes -> 0
    hs  = feat_dst + (long)row * DIM;
    na  = neigh    + (long)row * DIM;
  }
  int khalf = (lane >> 4) ? 8 : 0;

  v8f acc[8] = {};
  for (int kk = 0; kk < 8; ++kk) {
    const float* src   = (kk < 4) ? hs : na;
    float        scale = (kk < 4) ? 1.0f : inv;
    int kb = (kk & 3) * 32 + khalf;
    v16bf a = load_a_frag(src, kb, scale, valid);
    const __bf16* bbase = sW + (size_t)(kk * 256 + lane) * 16;
#pragma unroll
    for (int nt = 0; nt < 8; ++nt) {
      v16bf b = *(const v16bf*)(bbase + (size_t)nt * 512);
      acc[nt] = __builtin_amdgcn_wmma_f32_16x16x32_bf16(
          false, a, false, b, (short)0, acc[nt], false, false);
    }
  }

  int ncol  = lane & 15;
  int mbase = (lane >> 4) ? 8 : 0;
#pragma unroll
  for (int nt = 0; nt < 8; ++nt) {
    int j = nt * 16 + ncol;
    float bjv = bfc[j];
#pragma unroll
    for (int r = 0; r < 8; ++r) {
      long orow = r0 + mbase + r;
      if (orow < ndst)
        out[orow * DIM + j] = fmaxf(acc[nt][r] + bjv, 0.0f);
    }
  }
}

// ---------------------------------------------------------------------------
// Launcher. Inputs (setup_inputs order): feat_src, feat_dst, W1, b1, W2, b2,
// Wfc, bfc, edge_src, edge_dst.  Output: [N_DST, 128] f32.
// Workspace: 2x64KB packed weights + E f32 scores + 2*N_DST f32 + N_DST*128
// f32 neigh accumulator (~28 MB for the reference sizes).
// ---------------------------------------------------------------------------
extern "C" void kernel_launch(void* const* d_in, const int* in_sizes, int n_in,
                              void* d_out, int out_size, void* d_ws, size_t ws_size,
                              hipStream_t stream) {
  const float* feat_src = (const float*)d_in[0];
  const float* feat_dst = (const float*)d_in[1];
  const float* W1   = (const float*)d_in[2];
  const float* b1   = (const float*)d_in[3];
  const float* W2   = (const float*)d_in[4];
  const float* b2   = (const float*)d_in[5];
  const float* Wfc  = (const float*)d_in[6];
  const float* bfc  = (const float*)d_in[7];
  const int* edge_src = (const int*)d_in[8];
  const int* edge_dst = (const int*)d_in[9];

  const int E    = in_sizes[8];
  const int NDST = in_sizes[1] / DIM;

  char* ws = (char*)d_ws;
  size_t off = 0;
  auto take = [&](size_t bytes) -> void* {
    void* p = ws + off;
    off = (off + bytes + 255) & ~(size_t)255;
    return p;
  };
  __bf16* W1B   = (__bf16*)take(WFRAG_BYTES);
  __bf16* WfcB  = (__bf16*)take(WFRAG_BYTES);
  float*  score = (float*)take((size_t)E * 4);
  float*  smax  = (float*)take((size_t)NDST * 4);
  float*  denom = (float*)take((size_t)NDST * 4);
  float*  neigh = (float*)take((size_t)NDST * DIM * 4);
  (void)ws_size; (void)n_in; (void)out_size;

  // 0) pack weights into WMMA B-fragment layout (bf16)
  k_pack_weights<<<16, 256, 0, stream>>>(W1, Wfc, W1B, WfcB);

  // 1) init scratch
  {
    long total  = (long)NDST * (DIM + 2);
    int  blocks = (int)((total + 255) / 256);
    k_init<<<blocks, 256, 0, stream>>>(smax, denom, neigh, NDST);
  }

  // 2) edge MLP + score + segment max (fused, WMMA, persistent blocks)
  {
    int waves  = (E + 15) / 16;
    int blocks = (waves + 7) / 8;
    if (blocks > 1024) blocks = 1024;      // grid-stride; amortize LDS staging
    k_edge_scores<<<blocks, 256, 0, stream>>>(feat_src, feat_dst, edge_src,
                                              edge_dst, W1B, b1, W2, b2,
                                              score, smax, E);
  }

  // 3) exp + denom + weighted scatter (unnormalized)
  {
    int blocks = (E + 7) / 8;
    k_edge_scatter<<<blocks, 256, 0, stream>>>(feat_src, edge_src, edge_dst,
                                               score, smax, denom, neigh, E);
  }

  // 4) final GEMM + bias + relu (WMMA), normalization folded into A-load
  {
    int waves  = (NDST + 15) / 16;
    int blocks = (waves + 7) / 8;
    k_final<<<blocks, 256, 0, stream>>>(feat_dst, neigh, denom, WfcB, bfc,
                                        (float*)d_out, NDST);
  }
}
